// GATLayer_61392262529025
// MI455X (gfx1250) — matrix operand order repaired and checked by reading, not compiled, not run on hardware
//
#include <hip/hip_runtime.h>

#define HEADS 4
#define CH 64
#define HC 256
#define OUTD 64
#define SLOPE_ATT 0.2f
#define SLOPE_ACT 0.01f

typedef __attribute__((ext_vector_type(16))) __bf16 v16bf;
typedef __attribute__((ext_vector_type(8)))  float v8f;

// leaky_relu with slope in (0,1): max(v, s*v) -> single v_max_num_f32
__device__ __forceinline__ float lrelu(float v, float s) { return fmaxf(v, s * v); }

// order-preserving encode of f32 into u32 so atomicMax(u32) == float max
__device__ __forceinline__ unsigned encf(float f) {
  unsigned u = __float_as_uint(f);
  return (u & 0x80000000u) ? ~u : (u | 0x80000000u);
}
__device__ __forceinline__ float decf(unsigned u) {
  return __uint_as_float((u & 0x80000000u) ? (u ^ 0x80000000u) : ~u);
}

// s_h = dot(W[h*64:...], a_src[h,:]), t_h likewise. One block, 256 threads.
__global__ __launch_bounds__(256) void prep_st_kernel(
    const float* __restrict__ W, const float* __restrict__ asrc,
    const float* __restrict__ adst, float* __restrict__ st) {
  __shared__ float ps[256];
  __shared__ float pd[256];
  int t = threadIdx.x;
  float w = W[t];
  ps[t] = w * asrc[t];
  pd[t] = w * adst[t];
  __syncthreads();
  for (int off = 32; off >= 1; off >>= 1) {
    if ((t & 63) < off) { ps[t] += ps[t + off]; pd[t] += pd[t + off]; }
    __syncthreads();
  }
  if ((t & 63) == 0) {
    st[t >> 6] = ps[t];           // s_h
    st[4 + (t >> 6)] = pd[t];     // t_h
  }
}

// Pre-swizzle fcW [256,64] f32 -> bf16 WMMA B fragments.
// Layout: bfrag[(((nt*8 + kk)*32) + lane)*16 + e]
// B element (lane, e) of k-step kk, n-tile nt: k = kk*32 + (lane>=16 ? 8 : 0) + (e & 7) + (e >= 8 ? 16 : 0),
// col = nt*16 + (lane & 15).
__global__ __launch_bounds__(256) void prep_bfrag_kernel(
    const float* __restrict__ fcW, __bf16* __restrict__ bfrag) {
  int i = blockIdx.x * 256 + threadIdx.x;
  if (i >= 4 * 8 * 32 * 16) return;
  int e    = i & 15;
  int lane = (i >> 4) & 31;
  int kk   = (i >> 9) & 7;
  int nt   = (i >> 12) & 3;
  int k    = kk * 32 + ((lane >> 4) << 3) + (e & 7) + ((e >> 3) << 4);
  int col  = nt * 16 + (lane & 15);
  bfrag[i] = (__bf16)fcW[k * OUTD + col];
}

// init running max with the self-loop logit (self loops always exist -> finite max)
__global__ __launch_bounds__(256) void init_max_kernel(
    const float* __restrict__ x, const float* __restrict__ st,
    unsigned* __restrict__ mbits, int nh) {
  int i = blockIdx.x * 256 + threadIdx.x;
  if (i >= nh) return;
  int n = i >> 2, h = i & 3;
  float l = lrelu(x[n] * (st[h] + st[4 + h]), SLOPE_ATT);
  mbits[i] = encf(l);
}

__global__ __launch_bounds__(256) void edge_max_kernel(
    const float* __restrict__ x, const int* __restrict__ ei,
    const float* __restrict__ st, unsigned* __restrict__ mbits, int e) {
  int i = blockIdx.x * 256 + threadIdx.x;
  if (i >= e) return;
  int src = ei[i];
  int dst = ei[e + i];
  float xs = x[src], xd = x[dst];
#pragma unroll
  for (int h = 0; h < HEADS; ++h) {
    float l = lrelu(xs * st[h] + xd * st[4 + h], SLOPE_ATT);
    atomicMax(&mbits[dst * 4 + h], encf(l));
  }
}

// den = exp(self_logit - m); num = den * x[n]   (self-loop contribution)
__global__ __launch_bounds__(256) void init_sum_kernel(
    const float* __restrict__ x, const float* __restrict__ st,
    const unsigned* __restrict__ mbits, float* __restrict__ den,
    float* __restrict__ num, int nh) {
  int i = blockIdx.x * 256 + threadIdx.x;
  if (i >= nh) return;
  int n = i >> 2, h = i & 3;
  float xn = x[n];
  float l = lrelu(xn * (st[h] + st[4 + h]), SLOPE_ATT);
  float e0 = __expf(l - decf(mbits[i]));
  den[i] = e0;
  num[i] = e0 * xn;
}

__global__ __launch_bounds__(256) void edge_sum_kernel(
    const float* __restrict__ x, const int* __restrict__ ei,
    const float* __restrict__ st, const unsigned* __restrict__ mbits,
    float* __restrict__ den, float* __restrict__ num, int e) {
  int i = blockIdx.x * 256 + threadIdx.x;
  if (i >= e) return;
  int src = ei[i];
  int dst = ei[e + i];
  float xs = x[src], xd = x[dst];
#pragma unroll
  for (int h = 0; h < HEADS; ++h) {
    float l = lrelu(xs * st[h] + xd * st[4 + h], SLOPE_ATT);
    float ee = __expf(l - decf(mbits[dst * 4 + h]));
    atomicAdd(&den[dst * 4 + h], ee);
    atomicAdd(&num[dst * 4 + h], ee * xs);
  }
}

// Fused: r = num/(den+1e-16); h1 = leaky_relu(r_h*W[k] + b[k], 0.01) built
// directly as bf16 WMMA A fragments; out = h1 @ fcW + fcb via
// v_wmma_f32_16x16x32_bf16. One wave32 per 16-row tile; 8 waves/block.
__global__ __launch_bounds__(256) void node_fc_wmma_kernel(
    const float* __restrict__ num, const float* __restrict__ den,
    const float* __restrict__ W, const float* __restrict__ bias,
    const __bf16* __restrict__ bfrag, const float* __restrict__ fcb,
    float* __restrict__ out, int n) {
  __shared__ float sW[HC];
  __shared__ float sB[HC];
  int tid = threadIdx.x;
  sW[tid] = W[tid];
  sB[tid] = bias[tid];
  __syncthreads();

  int wave = tid >> 5;
  int lane = tid & 31;
  int ntiles = (n + 15) >> 4;
  int tile = blockIdx.x * 8 + wave;
  if (tile >= ntiles) return;          // whole-wave exit: EXEC stays all-ones

  int half = lane >> 4;
  int idx  = lane & 15;
  int row  = tile * 16 + idx;
  int rowL = row < n ? row : (n - 1);  // load clamp (N=50000 is exact, but be safe)

  float r4[HEADS];
#pragma unroll
  for (int h = 0; h < HEADS; ++h)
    r4[h] = num[rowL * 4 + h] / (den[rowL * 4 + h] + 1e-16f);

  // A fragments: 16-bit A 16x32 layout. lane<16: K 0..7 & 16..23; lane>=16: K 8..15 & 24..31.
  v16bf afrag[8];
#pragma unroll
  for (int kk = 0; kk < 8; ++kk) {
#pragma unroll
    for (int e = 0; e < 16; ++e) {
      int k = kk * 32 + half * 8 + (e & 7) + ((e >= 8) ? 16 : 0);
      float v = fmaf(r4[k >> 6], sW[k], sB[k]);
      v = fmaxf(v, SLOPE_ACT * v);     // leaky_relu as single max
      afrag[kk][e] = (__bf16)v;
    }
  }

#pragma unroll
  for (int nt = 0; nt < 4; ++nt) {
    v8f acc = {};
#pragma unroll
    for (int kk = 0; kk < 8; ++kk) {
      const v16bf bf =
          *(const v16bf*)(bfrag + (size_t)(((nt * 8 + kk) * 32) + lane) * 16);
      acc = __builtin_amdgcn_wmma_f32_16x16x32_bf16(
          false, afrag[kk], false, bf, (short)0, acc, false, false);
    }
    int col = nt * 16 + idx;
    float cb = fcb[col];
#pragma unroll
    for (int r = 0; r < 8; ++r) {
      int orow = tile * 16 + half * 8 + r;   // C/D layout: VGPR r -> M=r+8*half, N=idx
      if (orow < n) out[(size_t)orow * OUTD + col] = acc[r] + cb;
    }
  }
}

extern "C" void kernel_launch(void* const* d_in, const int* in_sizes, int n_in,
                              void* d_out, int out_size, void* d_ws, size_t ws_size,
                              hipStream_t stream) {
  (void)n_in; (void)out_size; (void)ws_size;
  const float* x1  = (const float*)d_in[0];
  const int*   ei1 = (const int*)d_in[1];
  const float* x2  = (const float*)d_in[3];
  const int*   ei2 = (const int*)d_in[4];
  const float* WA  = (const float*)d_in[6];
  const float* asA = (const float*)d_in[7];
  const float* adA = (const float*)d_in[8];
  const float* bA  = (const float*)d_in[9];
  const float* WB  = (const float*)d_in[10];
  const float* asB = (const float*)d_in[11];
  const float* adB = (const float*)d_in[12];
  const float* bB  = (const float*)d_in[13];
  const float* fcW = (const float*)d_in[14];
  const float* fcb = (const float*)d_in[15];
  float* out = (float*)d_out;

  int n = in_sizes[0];          // 50000
  int e = in_sizes[1] / 2;      // 400000

  char* ws = (char*)d_ws;
  size_t off = 0;
  size_t nhBytes = (((size_t)n * HEADS * 4) + 255) & ~(size_t)255;
  float*    st    = (float*)(ws + off);    off += 256;
  unsigned* mbits = (unsigned*)(ws + off); off += nhBytes;
  float*    den   = (float*)(ws + off);    off += nhBytes;
  float*    num   = (float*)(ws + off);    off += nhBytes;
  __bf16*   bfrag = (__bf16*)(ws + off);   off += 4 * 8 * 32 * 16 * 2;

  prep_bfrag_kernel<<<64, 256, 0, stream>>>(fcW, bfrag);

  int nh = n * HEADS;
  int nhBlocks   = (nh + 255) / 256;
  int eBlocks    = (e + 255) / 256;
  int ntiles     = (n + 15) / 16;
  int nodeBlocks = (ntiles + 7) / 8;

  for (int g = 0; g < 2; ++g) {
    const float* x   = g ? x2  : x1;
    const int*   ei  = g ? ei2 : ei1;
    const float* W   = g ? WB  : WA;
    const float* as_ = g ? asB : asA;
    const float* ad_ = g ? adB : adA;
    const float* b   = g ? bB  : bA;
    float* og = out + (size_t)g * n * OUTD;

    prep_st_kernel<<<1, 256, 0, stream>>>(W, as_, ad_, st);
    init_max_kernel<<<nhBlocks, 256, 0, stream>>>(x, st, mbits, nh);
    edge_max_kernel<<<eBlocks, 256, 0, stream>>>(x, ei, st, mbits, e);
    init_sum_kernel<<<nhBlocks, 256, 0, stream>>>(x, st, mbits, den, num, nh);
    edge_sum_kernel<<<eBlocks, 256, 0, stream>>>(x, ei, st, mbits, den, num, e);
    node_fc_wmma_kernel<<<nodeBlocks, 256, 0, stream>>>(num, den, W, b, bfrag,
                                                        fcb, og, n);
  }
}
